// GTLayer_1683627180463
// MI455X (gfx1250) — matrix operand order repaired and checked by reading, not compiled, not run on hardware
//
#include <hip/hip_runtime.h>

// Problem constants (match reference)
#define N_NODES 100000
#define E_EDGES 600000
#define LATDIM  128
#define HEAD    4
#define HDIM    32

typedef __attribute__((ext_vector_type(16))) _Float16 v16h;
typedef __attribute__((ext_vector_type(8)))  _Float16 v8h;
typedef __attribute__((ext_vector_type(4)))  _Float16 v4h;
typedef __attribute__((ext_vector_type(8)))  float    v8f;

__device__ __forceinline__ v16h load_frag16(const _Float16* p) {
    v8h lo = *(const v8h*)p;          // 16B aligned b128
    v8h hi = *(const v8h*)(p + 8);    // 16B aligned b128
    return __builtin_shufflevector(lo, hi, 0,1,2,3,4,5,6,7,8,9,10,11,12,13,14,15);
}

// ---------------------------------------------------------------------------
// Kernel 0: zero accumulation targets (harness poisons d_out / d_ws)
// ---------------------------------------------------------------------------
__global__ void zero_kernel(float* __restrict__ out, float* __restrict__ attNorm) {
    const int stride = gridDim.x * blockDim.x;
    const int tid    = blockIdx.x * blockDim.x + threadIdx.x;
    for (int i = tid; i < N_NODES * LATDIM; i += stride) out[i] = 0.0f;
    for (int i = tid; i < N_NODES * HEAD;   i += stride) attNorm[i] = 0.0f;
}

// ---------------------------------------------------------------------------
// Kernel P1: embeds f32 -> f16 row-major (vectorized: float4 in, 4xf16 out).
// A fragments then become contiguous b128 loads per lane.
// ---------------------------------------------------------------------------
__global__ void cvt_embeds_kernel(const float* __restrict__ embeds,
                                  _Float16* __restrict__ Ah) {
    const int stride = gridDim.x * blockDim.x;
    const int total  = (N_NODES * LATDIM) / 4;
    for (int i = blockIdx.x * blockDim.x + threadIdx.x; i < total; i += stride) {
        const float4 f = ((const float4*)embeds)[i];
        v4h o = { (_Float16)f.x, (_Float16)f.y, (_Float16)f.z, (_Float16)f.w };
        ((v4h*)Ah)[i] = o;
    }
}

// ---------------------------------------------------------------------------
// Kernel P2: pack q/k/v weights into fragment-major f16:
//   Wp[mat][ntile][kt][lane][i] = (f16) W[mat][(kt*32 + (lane>>4)*16 + i)*128
//                                             + ntile*16 + (lane&15)]
// so a wave's B fragment (contiguous k = 16*(lane>>4) + i, same order as A) is 32
// contiguous bytes per lane. 3*128*128 halfs = 96 KB total, L2-resident.
// ---------------------------------------------------------------------------
__global__ void pack_weights_kernel(const float* __restrict__ qW,
                                    const float* __restrict__ kW,
                                    const float* __restrict__ vW,
                                    _Float16* __restrict__ Wp) {
    const int idx = blockIdx.x * blockDim.x + threadIdx.x;  // [0, 3*16384)
    if (idx >= 3 * LATDIM * LATDIM) return;
    const int mat  = idx >> 14;
    const int rem  = idx & 16383;          // [ntile][kt][lane][i]
    const int i    = rem & 15;
    const int lane = (rem >> 4) & 31;
    const int kt   = (rem >> 9) & 3;
    const int nt   = rem >> 11;
    const int k = kt * 32 + (lane >> 4) * 16 + i;
    const int n = nt * 16 + (lane & 15);
    const float* W = (mat == 0) ? qW : (mat == 1) ? kW : vW;
    Wp[idx] = (_Float16)W[k * LATDIM + n];
}

// ---------------------------------------------------------------------------
// Kernel 1: node-level fused Q/K/V GEMM via v_wmma_f32_16x16x32_f16.
// Block = 128 threads (4 waves). Wave owns a 32(M)x16(N) strip: 2 M-tiles x
// 3 outputs = 6 accumulators; B fragments (2x b128 each) are reused across
// both M-tiles. 24 WMMAs per wave, all operands via b128 vector loads.
// Grid: (N/32, 2); blockIdx.y*4+wave covers the 8 N-tiles.
// ---------------------------------------------------------------------------
__global__ __launch_bounds__(128)
void qkv_wmma_kernel(const _Float16* __restrict__ Ah,
                     const _Float16* __restrict__ Wp,
                     float* __restrict__ Q,
                     float* __restrict__ Kd,
                     float* __restrict__ V) {
    const int lane  = threadIdx.x & 31;
    const int wave  = threadIdx.x >> 5;
    const int ntile = blockIdx.y * 4 + wave;
    const int m0    = blockIdx.x * 32;
    const int am    = lane & 15;       // A: M = lane%16
    const int ahalf = lane >> 4;       // A: lanes 16-31 -> K offset +16

    v8f c00 = {}, c01 = {}, c02 = {};  // M-tile 0: Q,K,V
    v8f c10 = {}, c11 = {}, c12 = {};  // M-tile 1: Q,K,V

    #pragma unroll
    for (int kt = 0; kt < 4; ++kt) {
        const size_t bo = ((size_t)(ntile * 4 + kt) * 32 + lane) * 16;
        const v16h bq = load_frag16(Wp + bo);
        const v16h bk = load_frag16(Wp + 16384 + bo);
        const v16h bv = load_frag16(Wp + 32768 + bo);

        const _Float16* a0p =
            Ah + (size_t)(m0 + am) * LATDIM + kt * 32 + ahalf * 16;
        // a[0..15]: K = kt*32 + ahalf*16 + {0..15}
        const v16h a0 = __builtin_shufflevector(
            *(const v8h*)a0p, *(const v8h*)(a0p + 8),
            0,1,2,3,4,5,6,7,8,9,10,11,12,13,14,15);
        const _Float16* a1p = a0p + 16 * LATDIM;
        const v16h a1 = __builtin_shufflevector(
            *(const v8h*)a1p, *(const v8h*)(a1p + 8),
            0,1,2,3,4,5,6,7,8,9,10,11,12,13,14,15);

        c00 = __builtin_amdgcn_wmma_f32_16x16x32_f16(false, a0, false, bq, (short)0, c00, false, false);
        c01 = __builtin_amdgcn_wmma_f32_16x16x32_f16(false, a0, false, bk, (short)0, c01, false, false);
        c02 = __builtin_amdgcn_wmma_f32_16x16x32_f16(false, a0, false, bv, (short)0, c02, false, false);
        c10 = __builtin_amdgcn_wmma_f32_16x16x32_f16(false, a1, false, bq, (short)0, c10, false, false);
        c11 = __builtin_amdgcn_wmma_f32_16x16x32_f16(false, a1, false, bk, (short)0, c11, false, false);
        c12 = __builtin_amdgcn_wmma_f32_16x16x32_f16(false, a1, false, bv, (short)0, c12, false, false);
    }

    // C/D f32 16x16 layout: VGPR j -> M = j + 8*(lane>=16), N = lane%16.
    const int cn = ntile * 16 + (lane & 15);
    #pragma unroll
    for (int j = 0; j < 8; ++j) {
        const size_t o0 = (size_t)(m0 + ahalf * 8 + j) * LATDIM + cn;
        Q[o0]  = c00[j];  Kd[o0] = c01[j];  V[o0]  = c02[j];
        const size_t o1 = o0 + (size_t)16 * LATDIM;
        Q[o1]  = c10[j];  Kd[o1] = c11[j];  V[o1]  = c12[j];
    }
}

// ---------------------------------------------------------------------------
// Kernel 2: per-edge attention logits. Each wave handles 2 edges; the second
// edge's gathered Q/K rows are prefetched (global_prefetch_b8) before the
// first edge is computed — indirect addresses no HW prefetcher can predict.
// Lane l covers dims 4l..4l+3 (head = l>>3); 3-step shuffle reduce per head.
// ---------------------------------------------------------------------------
__global__ __launch_bounds__(256)
void edge_logits_kernel(const float* __restrict__ Q,
                        const float* __restrict__ Kd,
                        const float* __restrict__ filt,
                        const int* __restrict__ rows,
                        const int* __restrict__ cols,
                        float* __restrict__ expAtt,
                        float* __restrict__ attNorm) {
    const int wv   = blockIdx.x * 8 + (threadIdx.x >> 5);   // (E/2) % 8 == 0
    const int lane = threadIdx.x & 31;
    const int e0   = wv * 2;

    int r[2], c[2];
    r[0] = rows[e0];     c[0] = cols[e0];
    r[1] = rows[e0 + 1]; c[1] = cols[e0 + 1];

    // prefetch edge-1 gather targets (each lane covers 16B -> full 512B row)
    __builtin_prefetch(Q  + (size_t)r[1] * LATDIM + lane * 4, 0, 3);
    __builtin_prefetch(Kd + (size_t)c[1] * LATDIM + lane * 4, 0, 3);

    #pragma unroll
    for (int t = 0; t < 2; ++t) {
        const int e = e0 + t;
        const float4 q4 = ((const float4*)(Q  + (size_t)r[t] * LATDIM))[lane];
        const float4 k4 = ((const float4*)(Kd + (size_t)c[t] * LATDIM))[lane];
        float p = q4.x * k4.x + q4.y * k4.y + q4.z * k4.z + q4.w * k4.w;

        p += __shfl_xor(p, 1, 32);
        p += __shfl_xor(p, 2, 32);
        p += __shfl_xor(p, 4, 32);

        if ((lane & 7) == 0) {
            const int h = lane >> 3;
            const float att =
                fminf(10.0f, fmaxf(-10.0f, p)) + filt[(size_t)c[t] * HEAD + h];
            const float ea = __expf(att);
            expAtt[(size_t)e * HEAD + h] = ea;
            unsafeAtomicAdd(&attNorm[(size_t)r[t] * HEAD + h], ea);
        }
    }
}

// ---------------------------------------------------------------------------
// Kernel 3: normalize + weighted-V scatter. 2 edges per wave with prefetch
// of the second edge's V row; 128 f32 atomic adds per edge into d_out.
// ---------------------------------------------------------------------------
__global__ __launch_bounds__(256)
void edge_scatter_kernel(const float* __restrict__ V,
                         const int* __restrict__ rows,
                         const int* __restrict__ cols,
                         const float* __restrict__ expAtt,
                         const float* __restrict__ attNorm,
                         float* __restrict__ out) {
    const int wv   = blockIdx.x * 8 + (threadIdx.x >> 5);
    const int lane = threadIdx.x & 31;
    const int e0   = wv * 2;
    const int h    = lane >> 3;        // dims 4*lane..4*lane+3 lie in head lane>>3

    int r[2], c[2];
    r[0] = rows[e0];     c[0] = cols[e0];
    r[1] = rows[e0 + 1]; c[1] = cols[e0 + 1];

    __builtin_prefetch(V + (size_t)c[1] * LATDIM + lane * 4, 0, 3);

    #pragma unroll
    for (int t = 0; t < 2; ++t) {
        const int e = e0 + t;
        const float ea   = expAtt[(size_t)e * HEAD + h];
        const float norm = attNorm[(size_t)r[t] * HEAD + h];
        const float w    = ea / (norm + 1e-8f);

        const float4 v4 = ((const float4*)(V + (size_t)c[t] * LATDIM))[lane];
        float* dst = out + (size_t)r[t] * LATDIM + lane * 4;
        unsafeAtomicAdd(dst + 0, w * v4.x);
        unsafeAtomicAdd(dst + 1, w * v4.y);
        unsafeAtomicAdd(dst + 2, w * v4.z);
        unsafeAtomicAdd(dst + 3, w * v4.w);
    }
}

// ---------------------------------------------------------------------------
// Host launcher
// ---------------------------------------------------------------------------
extern "C" void kernel_launch(void* const* d_in, const int* in_sizes, int n_in,
                              void* d_out, int out_size, void* d_ws, size_t ws_size,
                              hipStream_t stream) {
    const float* embeds = (const float*)d_in[0];
    const float* qW     = (const float*)d_in[1];
    const float* kW     = (const float*)d_in[2];
    const float* vW     = (const float*)d_in[3];
    const float* filt   = (const float*)d_in[4];
    const int*   rows   = (const int*)d_in[5];
    const int*   cols   = (const int*)d_in[6];
    float*       out    = (float*)d_out;

    // Workspace layout: Q | K | V | expAtt | attNorm (f32) | Ah | Wp (f16)
    float* Q       = (float*)d_ws;
    float* K       = Q + (size_t)N_NODES * LATDIM;
    float* V       = K + (size_t)N_NODES * LATDIM;
    float* expAtt  = V + (size_t)N_NODES * LATDIM;
    float* attNorm = expAtt + (size_t)E_EDGES * HEAD;
    _Float16* Ah   = (_Float16*)(attNorm + (size_t)N_NODES * HEAD);
    _Float16* Wp   = Ah + (size_t)N_NODES * LATDIM;

    zero_kernel<<<1024, 256, 0, stream>>>(out, attNorm);

    cvt_embeds_kernel<<<2048, 256, 0, stream>>>(embeds, Ah);
    pack_weights_kernel<<<(3 * LATDIM * LATDIM + 255) / 256, 256, 0, stream>>>(
        qW, kW, vW, Wp);

    dim3 gemmGrid(N_NODES / 32, 2);                        // 3125 x 2 blocks
    qkv_wmma_kernel<<<gemmGrid, 128, 0, stream>>>(Ah, Wp, Q, K, V);

    edge_logits_kernel<<<E_EDGES / 16, 256, 0, stream>>>(Q, K, filt, rows, cols,
                                                         expAtt, attNorm);

    edge_scatter_kernel<<<E_EDGES / 16, 256, 0, stream>>>(V, rows, cols,
                                                          expAtt, attNorm, out);
}
